// LiquidCellRef_65146063946241
// MI455X (gfx1250) — matrix operand order repaired and checked by reading, not compiled, not run on hardware
//
#include <hip/hip_runtime.h>
#include <hip/hip_bf16.h>
#include <math.h>

// Problem dims (match reference)
#define B_   8
#define T_   2048
#define H_   1024          // HID_D (N of GEMM)
#define K_   1024          // IN_D  (K of GEMM)
#define M_   (B_ * T_)     // 16384 GEMM rows
#define CT_  128           // scan chunk length
#define NC_  (T_ / CT_)    // 16 chunks

#define NBLK_    64        // N rows per block tile
#define PANEL_K  128       // K elements per TDM panel
#define PANELS   (K_ / PANEL_K)

typedef __attribute__((ext_vector_type(16))) __bf16 bf16x16;
typedef __attribute__((ext_vector_type(8)))  float  f32x8;
typedef __attribute__((ext_vector_type(4)))  unsigned int u32x4;
typedef __attribute__((ext_vector_type(8)))  int  i32x8;
typedef __attribute__((ext_vector_type(4)))  int  i32x4;

union Frag {
    bf16x16 v;
    uint4   q[2];   // 2 x 16B = 32B = 16 bf16
};

// --------------------------- TDM availability ------------------------------
#if defined(__has_builtin)
#  if __has_builtin(__builtin_amdgcn_tensor_load_to_lds) && \
      __has_builtin(__builtin_amdgcn_s_wait_tensorcnt)
#    define HAVE_TDM 1
#  endif
#endif
#ifndef HAVE_TDM
#  define HAVE_TDM 0
#endif

#if __has_include(<hip/amd_detail/amd_gfx1250_TDM.h>)
#  define TDM_SIX_ARGS 1   // therock / clang-23 lane: 6-arg builtin
#else
#  define TDM_SIX_ARGS 0   // ROCm 7.2 / clang-22 lane: 5-arg builtin
#endif

#if HAVE_TDM
__device__ __forceinline__ void tdm_issue(u32x4 g0, i32x8 g1) {
#if TDM_SIX_ARGS
    __builtin_amdgcn_tensor_load_to_lds(g0, g1, (i32x4)0, (i32x4)0, (i32x8)0, 0);
#else
    __builtin_amdgcn_tensor_load_to_lds(g0, g1, (i32x4)0, (i32x4)0, 0);
#endif
}

// Build a 2D D# that walks tile_dim1=64 rows x tile_dim0=PANEL_K cols of a
// row-major [H_, K_] bf16 matrix (row stride K_), landing row-major in LDS.
__device__ __forceinline__ void tdm_load_panel(const unsigned short* gsrc,
                                               unsigned lds_off) {
    unsigned long long ga = (unsigned long long)(uintptr_t)gsrc;
    u32x4 g0;
    g0[0] = 1u;                                    // count=1, user descriptor
    g0[1] = lds_off;                               // LDS byte address
    g0[2] = (unsigned)ga;                          // global_addr[31:0]
    g0[3] = (unsigned)(ga >> 32) | 0x80000000u;    // global_addr[56:32] | type=2
    i32x8 g1;
    g1[0] = (1 << 16);                             // data_size = 1 (2 bytes)
    g1[1] = (int)((K_ & 0xFFFF) << 16);            // tensor_dim0 lo16
    g1[2] = (int)((K_ >> 16) | ((H_ & 0xFFFF) << 16));  // dim0 hi | dim1 lo
    g1[3] = (int)((H_ >> 16) | (PANEL_K << 16));   // dim1 hi | tile_dim0
    g1[4] = NBLK_;                                 // tile_dim1 = 64 rows
    g1[5] = K_;                                    // tensor_dim0_stride lo32
    g1[6] = 0;                                     // stride0 hi | stride1 lo
    g1[7] = 0;
    tdm_issue(g0, g1);
}
#endif // HAVE_TDM

// ---------------------------------------------------------------------------
// fp32 -> bf16 (round-to-nearest-even), 4 elements per thread
// ---------------------------------------------------------------------------
__device__ __forceinline__ unsigned short f32_to_bf16_rne(float f) {
    unsigned int u = __float_as_uint(f);
    u += 0x7FFFu + ((u >> 16) & 1u);
    return (unsigned short)(u >> 16);
}

__global__ __launch_bounds__(256) void convert_bf16_kernel(
    const float* __restrict__ in, unsigned short* __restrict__ out, int n)
{
    int i = (blockIdx.x * 256 + threadIdx.x) * 4;
    if (i + 3 < n) {
        float4 v = *(const float4*)(in + i);
        ushort4 o;
        o.x = f32_to_bf16_rne(v.x);
        o.y = f32_to_bf16_rne(v.y);
        o.z = f32_to_bf16_rne(v.z);
        o.w = f32_to_bf16_rne(v.w);
        *(ushort4*)(out + i) = o;
    }
}

// ---------------------------------------------------------------------------
// Fused dual-GEMM + gating epilogue (TDM->LDS weight path when available).
//   accD = x @ Wd^T ; accB = x @ Wb^T     (bf16 WMMA, f32 accum)
//   delta = softplus(accD); At = exp(-delta*exp(A_log)); Bt = delta*accB
// Block: 256 thr = 8 waves, tile 128(M) x 64(N). Wave tile 32x32 (2x2 WMMA),
// both weight matrices share the A fragment -> 8 WMMAs / wave / K-step.
// Weights: TDM DMAs a 64x128 bf16 panel of each matrix into LDS, double-
// buffered; overlap via TENSORcnt (2 ops in flight beyond current buffer).
// ---------------------------------------------------------------------------
__global__ __launch_bounds__(256) void gemm_gate_kernel(
    const unsigned short* __restrict__ xb,   // [M_, K_] bf16
    const unsigned short* __restrict__ wdb,  // [H_, K_] bf16
    const unsigned short* __restrict__ wbb,  // [H_, K_] bf16
    const float* __restrict__ A_log,         // [H_]
    float* __restrict__ At,                  // [M_, H_]
    float* __restrict__ Bt)                  // [M_, H_]
{
    const int lane  = threadIdx.x & 31;
    const int wave  = threadIdx.x >> 5;   // 0..7
    const int waveM = wave & 3;           // 4 waves along M
    const int waveN = wave >> 2;          // 2 waves along N
    const int lM    = lane & 15;
    const int lHi   = lane >> 4;          // 0 / 1

    const int blockM = blockIdx.x * 128;
    const int blockN = blockIdx.y * NBLK_;

    const int mRow0 = blockM + waveM * 32 + lM;   // + i*16 per M-subtile
    const int nLoc0 = waveN * 32 + lM;            // local N row within block

#if HAVE_TDM
    __shared__ unsigned short ldsW[2][2][NBLK_ * PANEL_K];  // [buf][mat] 64 KB
#endif

    f32x8 accD[2][2];
    f32x8 accB[2][2];
#pragma unroll
    for (int i = 0; i < 2; ++i)
#pragma unroll
        for (int j = 0; j < 2; ++j) { accD[i][j] = (f32x8){}; accB[i][j] = (f32x8){}; }

#if HAVE_TDM
    if (wave == 0) {   // preload panel 0 into buffer 0
        tdm_load_panel(wdb + (size_t)blockN * K_,
                       (unsigned)(uintptr_t)&ldsW[0][0][0]);
        tdm_load_panel(wbb + (size_t)blockN * K_,
                       (unsigned)(uintptr_t)&ldsW[0][1][0]);
    }
#endif

    for (int p = 0; p < PANELS; ++p) {
#if HAVE_TDM
        const int buf = p & 1;
        if (wave == 0) {
            if (p + 1 < PANELS) {   // prefetch next panel into other buffer
                const size_t goff = (size_t)blockN * K_ + (size_t)(p + 1) * PANEL_K;
                tdm_load_panel(wdb + goff, (unsigned)(uintptr_t)&ldsW[buf ^ 1][0][0]);
                tdm_load_panel(wbb + goff, (unsigned)(uintptr_t)&ldsW[buf ^ 1][1][0]);
                __builtin_amdgcn_s_wait_tensorcnt(2);  // current buffer landed
            } else {
                __builtin_amdgcn_s_wait_tensorcnt(0);
            }
        }
        __syncthreads();           // LDS panel visible to all waves
#endif

#pragma unroll
        for (int k4 = 0; k4 < PANEL_K / 32; ++k4) {
            const int k = p * PANEL_K + k4 * 32;

            // A fragment, contiguous k order (same as B): lane<16 holds K {0..15},
            // lane>=16 holds K {16..31} -> two 16B runs at +16*lHi, +8.
            const int kA = k + lHi * 16;
            Frag a[2];
#pragma unroll
            for (int i = 0; i < 2; ++i) {
                const unsigned short* pa = xb + (size_t)(mRow0 + i * 16) * K_ + kA;
                a[i].q[0] = *(const uint4*)(pa);
                a[i].q[1] = *(const uint4*)(pa + 8);
            }

            // B fragment, 32x16 layout: lane holds 16 contiguous K at
            // +16*lHi, column n = lane&15 (row n of W == column n of B).
            Frag bD[2], bB[2];
#if HAVE_TDM
            const int kc = k4 * 32 + lHi * 16;     // column within panel
#pragma unroll
            for (int j = 0; j < 2; ++j) {
                const unsigned short* pd = &ldsW[buf][0][(nLoc0 + j * 16) * PANEL_K + kc];
                bD[j].q[0] = *(const uint4*)(pd);
                bD[j].q[1] = *(const uint4*)(pd + 8);
                const unsigned short* pb = &ldsW[buf][1][(nLoc0 + j * 16) * PANEL_K + kc];
                bB[j].q[0] = *(const uint4*)(pb);
                bB[j].q[1] = *(const uint4*)(pb + 8);
            }
#else
            const int kB = k + lHi * 16;
#pragma unroll
            for (int j = 0; j < 2; ++j) {
                const unsigned short* pd = wdb + (size_t)(blockN + nLoc0 + j * 16) * K_ + kB;
                bD[j].q[0] = *(const uint4*)(pd);
                bD[j].q[1] = *(const uint4*)(pd + 8);
                const unsigned short* pb = wbb + (size_t)(blockN + nLoc0 + j * 16) * K_ + kB;
                bB[j].q[0] = *(const uint4*)(pb);
                bB[j].q[1] = *(const uint4*)(pb + 8);
            }
#endif

#pragma unroll
            for (int i = 0; i < 2; ++i)
#pragma unroll
                for (int j = 0; j < 2; ++j) {
                    accD[i][j] = __builtin_amdgcn_wmma_f32_16x16x32_bf16(
                        false, a[i].v, false, bD[j].v, (short)0, accD[i][j], false, false);
                    accB[i][j] = __builtin_amdgcn_wmma_f32_16x16x32_bf16(
                        false, a[i].v, false, bB[j].v, (short)0, accB[i][j], false, false);
                }
        }

#if HAVE_TDM
        __syncthreads();  // reads done before this buffer is overwritten
#endif
    }

    // Epilogue. C/D layout: VGPR r, lane L: M = r + 8*(L>>4), N = L&15.
#pragma unroll
    for (int j = 0; j < 2; ++j) {
        const int n  = blockN + waveN * 32 + j * 16 + lM;
        const float ea = __expf(A_log[n]);
#pragma unroll
        for (int i = 0; i < 2; ++i) {
#pragma unroll
            for (int r = 0; r < 8; ++r) {
                const int m = blockM + waveM * 32 + i * 16 + r + lHi * 8;
                const float d     = accD[i][j][r];
                const float delta = (d > 20.0f) ? d : log1pf(__expf(d));
                const float av    = __expf(-delta * ea);
                const float bv    = delta * accB[i][j][r];
                const size_t idx  = (size_t)m * H_ + n;
                At[idx] = av;
                Bt[idx] = bv;
            }
        }
    }
}

// ---------------------------------------------------------------------------
// Chunked parallel scan of h = a*h + b over T.
// Pass A: per (b, chunk, h) compute chunk affine summary (P, Q).
// ---------------------------------------------------------------------------
__global__ __launch_bounds__(256) void scan_reduce_kernel(
    const float* __restrict__ At, const float* __restrict__ Bt,
    float* __restrict__ Ap, float* __restrict__ Bc)
{
    const int tid = blockIdx.x * 256 + threadIdx.x;   // B_*NC_*H_ threads
    const int h = tid & (H_ - 1);
    const int c = (tid >> 10) & (NC_ - 1);
    const int b = tid >> 14;

    size_t idx = ((size_t)(b * T_ + c * CT_)) * H_ + h;
    float P = 1.0f, Q = 0.0f;
#pragma unroll 4
    for (int t = 0; t < CT_; ++t) {
        const float a = At[idx];
        const float v = Bt[idx];
        Q = fmaf(a, Q, v);
        P = a * P;
        idx += H_;
    }
    const size_t s = ((size_t)(b * NC_ + c) << 10) + h;
    Ap[s] = P;
    Bc[s] = Q;
}

// Pass B: per (b,h) chain the NC_ chunk summaries; store h at each chunk start.
__global__ __launch_bounds__(256) void scan_chain_kernel(
    const float* __restrict__ Ap, const float* __restrict__ Bc,
    float* __restrict__ Hs)
{
    const int tid = blockIdx.x * 256 + threadIdx.x;   // B_*H_ threads
    const int h = tid & (H_ - 1);
    const int b = tid >> 10;

    float hst = 0.0f;
#pragma unroll
    for (int c = 0; c < NC_; ++c) {
        const size_t s = ((size_t)(b * NC_ + c) << 10) + h;
        Hs[s] = hst;
        hst = fmaf(Ap[s], hst, Bc[s]);
    }
}

// Pass C: per (b, chunk, h) replay the chunk from its start state, emit tanh.
__global__ __launch_bounds__(256) void scan_apply_kernel(
    const float* __restrict__ At, const float* __restrict__ Bt,
    const float* __restrict__ Hs, float* __restrict__ out)
{
    const int tid = blockIdx.x * 256 + threadIdx.x;   // B_*NC_*H_ threads
    const int h = tid & (H_ - 1);
    const int c = (tid >> 10) & (NC_ - 1);
    const int b = tid >> 14;

    const size_t s = ((size_t)(b * NC_ + c) << 10) + h;
    float hst = Hs[s];

    size_t idx = ((size_t)(b * T_ + c * CT_)) * H_ + h;
#pragma unroll 4
    for (int t = 0; t < CT_; ++t) {
        hst = fmaf(At[idx], hst, Bt[idx]);
        out[idx] = tanhf(hst);
        idx += H_;
    }
}

// ---------------------------------------------------------------------------
extern "C" void kernel_launch(void* const* d_in, const int* in_sizes, int n_in,
                              void* d_out, int out_size, void* d_ws, size_t ws_size,
                              hipStream_t stream)
{
    const float* x     = (const float*)d_in[0];   // [B_,T_,K_]
    const float* Wd    = (const float*)d_in[1];   // [H_,K_]
    const float* Wb    = (const float*)d_in[2];   // [H_,K_]
    const float* A_log = (const float*)d_in[3];   // [H_]
    float* out = (float*)d_out;                   // [B_,T_,H_]

    char* ws = (char*)d_ws;
    unsigned short* xb  = (unsigned short*)ws;  ws += (size_t)M_ * K_ * sizeof(unsigned short);
    unsigned short* wdb = (unsigned short*)ws;  ws += (size_t)H_ * K_ * sizeof(unsigned short);
    unsigned short* wbb = (unsigned short*)ws;  ws += (size_t)H_ * K_ * sizeof(unsigned short);
    float* At = (float*)ws;                     ws += (size_t)M_ * H_ * sizeof(float);
    float* Bt = (float*)ws;                     ws += (size_t)M_ * H_ * sizeof(float);
    float* Ap = (float*)ws;                     ws += (size_t)B_ * NC_ * H_ * sizeof(float);
    float* Bc = (float*)ws;                     ws += (size_t)B_ * NC_ * H_ * sizeof(float);
    float* Hs = (float*)ws;                     ws += (size_t)B_ * NC_ * H_ * sizeof(float);

    // 1) fp32 -> bf16
    {
        const int nx = M_ * K_;
        convert_bf16_kernel<<<nx / (256 * 4), 256, 0, stream>>>(x, xb, nx);
        const int nw = H_ * K_;
        convert_bf16_kernel<<<nw / (256 * 4), 256, 0, stream>>>(Wd, wdb, nw);
        convert_bf16_kernel<<<nw / (256 * 4), 256, 0, stream>>>(Wb, wbb, nw);
    }

    // 2) fused dual-GEMM + gate epilogue -> At, Bt (L2-resident, 128 MB)
    {
        dim3 grid(M_ / 128, H_ / NBLK_);
        gemm_gate_kernel<<<grid, 256, 0, stream>>>(xb, wdb, wbb, A_log, At, Bt);
    }

    // 3) chunked scan + tanh readout
    {
        const int nA = B_ * NC_ * H_;                    // 131072
        scan_reduce_kernel<<<nA / 256, 256, 0, stream>>>(At, Bt, Ap, Bc);
        const int nB = B_ * H_;                          // 8192
        scan_chain_kernel<<<nB / 256, 256, 0, stream>>>(Ap, Bc, Hs);
        scan_apply_kernel<<<nA / 256, 256, 0, stream>>>(At, Bt, Hs, out);
    }
}
